// TransformerChannelAttention_10067403342534
// MI455X (gfx1250) — matrix operand order repaired and checked by reading, not compiled, not run on hardware
//
#include <hip/hip_runtime.h>

#define BATCH  16
#define CH     256
#define NSP    9216            // 96*96
#define NCHUNK 16
#define CHUNK  (NSP / NCHUNK)  // 576 columns per workgroup
#define KSTEP  32
#define NKS    (CHUNK / KSTEP) // 18 k-steps
#define HEADS  8
#define HD     32
#define LN_EPS 1e-5f

typedef __attribute__((ext_vector_type(16))) __bf16 v16bf;
typedef __attribute__((ext_vector_type(8)))  float  v8f;

struct alignas(16) U128 { unsigned int x, y, z, w; };
struct alignas(16) F128 { float x, y, z, w; };

union FragU { v16bf bf; U128 q[2]; };

__device__ __forceinline__ unsigned short f2bf(float f) {
  unsigned u = __float_as_uint(f);
  u += 0x7FFFu + ((u >> 16) & 1u);          // round-to-nearest-even
  return (unsigned short)(u >> 16);
}

__device__ __forceinline__ void atomicAddF(float* p, float v) {
  unsafeAtomicAdd(p, v);                    // global_atomic_add_f32
}

// async global -> LDS, 16B per lane; INST_OFFSET applies to BOTH the LDS and
// the memory address (ISA 08_async_tensor §4.4), tracked by ASYNCcnt.
#define ASYNC_LD16(OFF)                                                      \
  asm volatile("global_load_async_to_lds_b128 %0, %1, %2 offset:" #OFF      \
               :: "v"(ldsb), "v"(voff), "s"(gbase) : "memory")

// ---------------------------------------------------------------------------
// Kernel 1: fused LayerNorm(channel) -> bf16 -> Gram G[b] = Xn Xn^T via WMMA,
// plus per-channel sum of xn (for mean_n(v)). One WG = one (batch, n-chunk).
// x tile is filled with GLOBAL_LOAD_ASYNC_TO_LDS_B128 (CDNA5 async path).
// ---------------------------------------------------------------------------
__global__ __launch_bounds__(256, 1) void k_ln_gram(
    const float* __restrict__ x, const float* __restrict__ lnw,
    const float* __restrict__ lnb, float* __restrict__ G,
    float* __restrict__ xnsum)
{
  __shared__ __align__(16) float xs[CH][36];             // f32 tile, 144B rows
  __shared__ __align__(16) unsigned short xbh[CH][40];   // bf16 tile, 80B rows
  __shared__ float psum[8][32];
  __shared__ float psq[8][32];
  __shared__ float cmu[32];
  __shared__ float crs[32];

  const int tid  = threadIdx.x;        // == channel index
  const int lane = tid & 31;
  const int wid  = tid >> 5;           // 8 waves
  const int b     = blockIdx.y;
  const int chunk = blockIdx.x;

  // Uniform SGPR base for GVS-mode async loads; per-thread part goes in voff.
  const float* gbase = x + ((size_t)b * CH) * NSP + (size_t)chunk * CHUNK;
  const unsigned ldsb = (unsigned)(uintptr_t)&xs[tid][0];  // LDS byte offset
  const float w_ln = lnw[tid];
  const float b_ln = lnb[tid];

  // Each wave owns row-blocks {2*wid, 2*wid+1} x all 16 col-blocks.
  v8f acc[2][16];
  const v8f vz = {0.f, 0.f, 0.f, 0.f, 0.f, 0.f, 0.f, 0.f};
#pragma unroll
  for (int i = 0; i < 2; ++i)
#pragma unroll
    for (int j = 0; j < 16; ++j) acc[i][j] = vz;

  float xsum_local = 0.f;

  for (int ks = 0; ks < NKS; ++ks) {
    // ---- async-fill the f32 tile: 32 floats (128B) for this channel -------
    const unsigned voff = (unsigned)tid * (NSP * 4u) + (unsigned)ks * (KSTEP * 4u);
    ASYNC_LD16(0);  ASYNC_LD16(16); ASYNC_LD16(32);  ASYNC_LD16(48);
    ASYNC_LD16(64); ASYNC_LD16(80); ASYNC_LD16(96);  ASYNC_LD16(112);
    if (ks + 1 < NKS)
      __builtin_prefetch(gbase + (size_t)tid * NSP + (ks + 1) * KSTEP, 0, 0);
    asm volatile("s_wait_asynccnt 0x0" ::: "memory");
    __syncthreads();

    // ---- column stats over C=256 (segmented reduce) -----------------------
    {
      const int col = tid & 31, seg = tid >> 5;
      float s = 0.f, s2 = 0.f;
#pragma unroll 8
      for (int r = 0; r < 32; ++r) {
        float v = xs[seg * 32 + r][col];
        s += v; s2 += v * v;
      }
      psum[seg][col] = s; psq[seg][col] = s2;
    }
    __syncthreads();
    if (tid < 32) {
      float S = 0.f, S2 = 0.f;
#pragma unroll
      for (int g = 0; g < 8; ++g) { S += psum[g][tid]; S2 += psq[g][tid]; }
      float mu  = S * (1.f / CH);
      float var = S2 * (1.f / CH) - mu * mu;
      cmu[tid] = mu;
      crs[tid] = rsqrtf(var + LN_EPS);
    }
    __syncthreads();

    // ---- normalize, accumulate column-sum, pack bf16 tile -----------------
    {
      union { F128 q[8]; float s[32]; } rx;
      const F128* xr = (const F128*)&xs[tid][0];
#pragma unroll
      for (int i = 0; i < 8; ++i) rx.q[i] = xr[i];
      unsigned* xpk = (unsigned*)&xbh[tid][0];
#pragma unroll
      for (int j = 0; j < 16; ++j) {
        float v0 = (rx.s[2 * j]     - cmu[2 * j])     * crs[2 * j]     * w_ln + b_ln;
        float v1 = (rx.s[2 * j + 1] - cmu[2 * j + 1]) * crs[2 * j + 1] * w_ln + b_ln;
        xsum_local += v0 + v1;
        xpk[j] = (unsigned)f2bf(v0) | ((unsigned)f2bf(v1) << 16);
      }
    }
    __syncthreads();

    // ---- WMMA: G-tile accumulation ---------------------------------------
    // A frag (16x32 bf16): lane<16 -> row M=lane, K runs {0..7,16..23};
    //                      lane>=16 -> row M=lane-16, K runs {8..15,24..31}.
    FragU fa0, fa1;
    const int arow = (2 * wid) * 16 + (lane & 15);
    const int akh  = (lane >> 4) * 8;
    fa0.q[0] = *(const U128*)&xbh[arow][akh];
    fa0.q[1] = *(const U128*)&xbh[arow][16 + akh];
    fa1.q[0] = *(const U128*)&xbh[arow + 16][akh];
    fa1.q[1] = *(const U128*)&xbh[arow + 16][16 + akh];

    // B frag (32x16 bf16): lane<16 -> N=lane, K runs {0..7,16..23}; lane>=16 -> K runs {8..15,24..31}.
    const int bkoff = (lane >> 4) * 8;
#pragma unroll
    for (int cb = 0; cb < 16; ++cb) {
      FragU fb;
      const int crow = cb * 16 + (lane & 15);
      fb.q[0] = *(const U128*)&xbh[crow][bkoff];
      fb.q[1] = *(const U128*)&xbh[crow][bkoff + 16];
      acc[0][cb] = __builtin_amdgcn_wmma_f32_16x16x32_bf16(
          false, fa0.bf, false, fb.bf, (short)0, acc[0][cb], false, false);
      acc[1][cb] = __builtin_amdgcn_wmma_f32_16x16x32_bf16(
          false, fa1.bf, false, fb.bf, (short)0, acc[1][cb], false, false);
    }
    __syncthreads();
  }

  // ---- epilogue: atomic-accumulate partial Gram + column sums -------------
  float* Gb = G + (size_t)b * CH * CH;
  const int nc   = lane & 15;
  const int mofs = (lane >> 4) * 8;   // C/D layout: VGPR v -> M = v + 8*(lane/16)
#pragma unroll
  for (int i = 0; i < 2; ++i) {
    const int rbase = (2 * wid + i) * 16 + mofs;
#pragma unroll
    for (int cb = 0; cb < 16; ++cb) {
      float* gp = Gb + (size_t)rbase * CH + cb * 16 + nc;
#pragma unroll
      for (int v = 0; v < 8; ++v)
        atomicAddF(gp + (size_t)v * CH, acc[i][cb][v]);
    }
  }
  atomicAddF(&xnsum[b * CH + tid], xsum_local);
}

// ---------------------------------------------------------------------------
// Kernel 2: per (b,h): S = Wq G Wk^T, softmax over e, vmean = Wv xnsum / n,
// va = attn @ vmean. Grid = B*HEADS blocks of 256.
// ---------------------------------------------------------------------------
__global__ __launch_bounds__(256) void k_attn(
    const float* __restrict__ wqkv, const float* __restrict__ G,
    const float* __restrict__ xnsum, float* __restrict__ va)
{
  __shared__ float bufA[CH * HD];   // phase 1: WkT[c2][e]; later Sl/vmean/xsl
  __shared__ float Tl[CH * HD];     // Tl[c][e] = (G Wk^T)[c][e]

  const int tid = threadIdx.x;
  const int b = blockIdx.x >> 3;
  const int h = blockIdx.x & 7;

  // stage WkT[c2=tid][e] (k rows live at wqkv rows [CH, 2CH))
#pragma unroll
  for (int e = 0; e < HD; ++e)
    bufA[tid * HD + e] = wqkv[(size_t)(CH + h * HD + e) * CH + tid];
  __syncthreads();

  // T[c=tid][e] = sum_c2 G[b][tid][c2] * Wk[e][c2]
  {
    const float* Gr = G + ((size_t)b * CH + tid) * CH;
    float tacc[HD];
#pragma unroll
    for (int e = 0; e < HD; ++e) tacc[e] = 0.f;
    for (int c2 = 0; c2 < CH; ++c2) {
      float g = Gr[c2];
      const F128* wk4 = (const F128*)&bufA[c2 * HD];
#pragma unroll
      for (int e4 = 0; e4 < 8; ++e4) {
        F128 wv = wk4[e4];
        tacc[e4 * 4 + 0] += g * wv.x;
        tacc[e4 * 4 + 1] += g * wv.y;
        tacc[e4 * 4 + 2] += g * wv.z;
        tacc[e4 * 4 + 3] += g * wv.w;
      }
    }
#pragma unroll
    for (int e = 0; e < HD; ++e) Tl[tid * HD + e] = tacc[e];
  }
  __syncthreads();

  float* Sl    = bufA;                 // [32][33] (WkT is dead now)
  float* vmean = bufA + 32 * 33;       // [32]
  float* xsl   = bufA + 32 * 33 + 32;  // [256]

  // S[d][e]: thread t -> d = t>>3, e-group = (t&7)*4
  {
    const int d  = tid >> 3;
    const int eg = (tid & 7) * 4;
    const float* wq = wqkv + (size_t)(h * HD + d) * CH;  // q rows at [0, CH)
    float s0 = 0.f, s1 = 0.f, s2 = 0.f, s3 = 0.f;
    for (int c = 0; c < CH; ++c) {
      float q = wq[c];
      F128 tv = *(const F128*)&Tl[c * HD + eg];
      s0 += q * tv.x; s1 += q * tv.y; s2 += q * tv.z; s3 += q * tv.w;
    }
    Sl[d * 33 + eg + 0] = s0;
    Sl[d * 33 + eg + 1] = s1;
    Sl[d * 33 + eg + 2] = s2;
    Sl[d * 33 + eg + 3] = s3;
  }
  xsl[tid] = xnsum[b * CH + tid];
  __syncthreads();

  // softmax over e (scaled), row d = tid
  if (tid < 32) {
    const float scale = 0.17677669529663687f;  // 32^-0.5
    float row[32];
    float m = -3.4e38f;
#pragma unroll
    for (int e = 0; e < 32; ++e) { row[e] = Sl[tid * 33 + e]; m = fmaxf(m, row[e]); }
    float s = 0.f;
#pragma unroll
    for (int e = 0; e < 32; ++e) { row[e] = __expf((row[e] - m) * scale); s += row[e]; }
    float inv = 1.f / s;
#pragma unroll
    for (int e = 0; e < 32; ++e) Sl[tid * 33 + e] = row[e] * inv;
  }

  // vmean[e=tid] = Wv[e] . xnsum[b] / n   (v rows at [2CH, 3CH))
  if (tid < 32) {
    const float* wv = wqkv + (size_t)(2 * CH + h * HD + tid) * CH;
    float a = 0.f;
    for (int c = 0; c < CH; ++c) a += wv[c] * xsl[c];
    vmean[tid] = a * (1.f / (float)NSP);
  }
  __syncthreads();

  // va[d=tid] = sum_e attn[d][e] * vmean[e]
  if (tid < 32) {
    float a = 0.f;
#pragma unroll
    for (int e = 0; e < 32; ++e) a += Sl[tid * 33 + e] * vmean[e];
    va[(size_t)b * CH + h * HD + tid] = a;
  }
}

// ---------------------------------------------------------------------------
// Kernel 3: out[b][c] = sigmoid(W_proj[c] . va[b] + b_proj[c])
// ---------------------------------------------------------------------------
__global__ __launch_bounds__(256) void k_proj(
    const float* __restrict__ wproj, const float* __restrict__ bproj,
    const float* __restrict__ va, float* __restrict__ out)
{
  __shared__ float vs[CH];
  const int b = blockIdx.x;
  const int c = threadIdx.x;
  vs[c] = va[(size_t)b * CH + c];
  __syncthreads();
  float acc = bproj[c];
  const float* wr = wproj + (size_t)c * CH;
  for (int i = 0; i < CH; ++i) acc += wr[i] * vs[i];
  out[(size_t)b * CH + c] = 1.f / (1.f + __expf(-acc));
}

// ---------------------------------------------------------------------------
extern "C" void kernel_launch(void* const* d_in, const int* in_sizes, int n_in,
                              void* d_out, int out_size, void* d_ws, size_t ws_size,
                              hipStream_t stream) {
  const float* x     = (const float*)d_in[0];
  const float* lnw   = (const float*)d_in[1];
  const float* lnb   = (const float*)d_in[2];
  const float* wqkv  = (const float*)d_in[3];
  const float* wproj = (const float*)d_in[4];
  const float* bproj = (const float*)d_in[5];
  (void)in_sizes; (void)n_in; (void)out_size; (void)ws_size;

  float* G     = (float*)d_ws;                      // B*256*256 f32 = 4 MB
  float* xnsum = G + (size_t)BATCH * CH * CH;       // B*256 f32
  float* va    = xnsum + (size_t)BATCH * CH;        // B*256 f32

  // zero the atomic accumulators (graph-capture safe)
  hipMemsetAsync(d_ws, 0,
                 ((size_t)BATCH * CH * CH + (size_t)BATCH * CH) * sizeof(float),
                 stream);

  k_ln_gram<<<dim3(NCHUNK, BATCH), 256, 0, stream>>>(x, lnw, lnb, G, xnsum);
  k_attn<<<BATCH * HEADS, 256, 0, stream>>>(wqkv, G, xnsum, va);
  k_proj<<<BATCH, 256, 0, stream>>>(wproj, bproj, va, (float*)d_out);
}
